// FCOSModule_7464653160912
// MI455X (gfx1250) — compile-verified, matrix operand order checked by reading, not run on hardware
//
#include <hip/hip_runtime.h>
#include <hip/hip_bf16.h>
#include <math.h>

// ---------------------------------------------------------------------------
// Types for CDNA5 WMMA (wave32, v_wmma_f32_16x16x32_bf16)
// ---------------------------------------------------------------------------
typedef __attribute__((ext_vector_type(16))) __bf16 v16bf;
typedef __attribute__((ext_vector_type(8)))  float  v8f;

union FragU {
    v16bf v;
    uint4 q[2];
};

union OutPack {
    uint4  q;
    __bf16 b[8];
};

// ---------------------------------------------------------------------------
// BN-fold + repack: W[Cout][Cin][KT] f32  ->  Wf[KT][Cout][Cin] bf16 (scaled)
// biasf[co] = s*b + beta - m*s  with  s = g * rsqrt(v + eps)
// ---------------------------------------------------------------------------
__global__ void fcos_fold_kernel(const float* __restrict__ W,
                                 const float* __restrict__ b,
                                 const float* __restrict__ g,
                                 const float* __restrict__ beta,
                                 const float* __restrict__ m,
                                 const float* __restrict__ vv,
                                 __bf16* __restrict__ Wf,
                                 float* __restrict__ biasf,
                                 int Cout, int Cin, int KT)
{
    int idx = blockIdx.x * blockDim.x + threadIdx.x;
    if (idx >= Cout * Cin) return;
    int co = idx / Cin;
    int ci = idx - co * Cin;
    float s = g[co] * rsqrtf(vv[co] + 1e-5f);
    for (int tap = 0; tap < KT; ++tap) {
        float w = W[((size_t)co * Cin + ci) * KT + tap] * s;
        Wf[((size_t)tap * Cout + co) * Cin + ci] = (__bf16)w;
    }
    if (ci == 0) biasf[co] = b[co] * s + beta[co] - m[co] * s;
}

// ---------------------------------------------------------------------------
// Transpose+cast: x f32 [B][C][T] -> xb bf16 [B][T][C]  (32x32 LDS tiles)
// ---------------------------------------------------------------------------
__global__ void fcos_transpose_cast(const float* __restrict__ x,
                                    __bf16* __restrict__ xb,
                                    int C, int T)
{
    __shared__ float tile[32][33];
    int b  = blockIdx.z;
    int c0 = blockIdx.y * 32;
    int t0 = blockIdx.x * 32;
    int tx = threadIdx.x;      // 0..31 (along t for reads)
    int ty = threadIdx.y;      // 0..7
    const float* xp = x + ((size_t)b * C) * T;
#pragma unroll
    for (int j = 0; j < 4; ++j)
        tile[ty + 8 * j][tx] = xp[(size_t)(c0 + ty + 8 * j) * T + (t0 + tx)];
    __syncthreads();
    __bf16* yp = xb + ((size_t)b * T) * C;
#pragma unroll
    for (int j = 0; j < 4; ++j)
        yp[(size_t)(t0 + ty + 8 * j) * C + (c0 + tx)] = (__bf16)tile[tx][ty + 8 * j];
}

// ---------------------------------------------------------------------------
// Main compute: Conv1d(K=1 or 3) as WMMA GEMM, bf16 in / bf16 out, f32 accum.
//   X : [B][T][cinRow] channels-last bf16, read at channel offset cinOff
//   W : [KT][Cout][Cin] bf16 (BN-folded)
//   Y : [B][T][coutRow] bf16, written at channel offset coutOff (+bias, ReLU)
// One wave computes a 64(M=cout) x 32(N=t) tile: 8 accumulators.
// Grid: x = group of 8 t-tiles (one per wave), y = cout block, z = batch.
// ---------------------------------------------------------------------------
__global__ __launch_bounds__(256)
void fcos_conv_wmma(const __bf16* __restrict__ X, int cinRow, int cinOff,
                    const __bf16* __restrict__ W,
                    const float* __restrict__ bias,
                    __bf16* __restrict__ Y, int coutRow, int coutOff,
                    int Cin, int Cout, int KT, int T, int relu)
{
    const int lane  = threadIdx.x & 31;
    const int wib   = threadIdx.x >> 5;          // wave in block (0..7)
    const int tbase = (blockIdx.x * 8 + wib) * 32;
    const int mbase = blockIdx.y * 64;
    const int b     = blockIdx.z;

    const int ln15  = lane & 15;
    const int khalf = (lane < 16) ? 0 : 8;       // 16-bit fragment K-group select
    const int kc    = KT >> 1;                   // center tap

    const v8f zacc = {0.f, 0.f, 0.f, 0.f, 0.f, 0.f, 0.f, 0.f};
    v8f acc[4][2];
#pragma unroll
    for (int mf = 0; mf < 4; ++mf)
#pragma unroll
        for (int nf = 0; nf < 2; ++nf)
            acc[mf][nf] = zacc;

    const __bf16* Xb = X + (size_t)b * T * cinRow + cinOff;
    const uint4 z4 = make_uint4(0u, 0u, 0u, 0u);

    for (int c0 = 0; c0 < Cin; c0 += 32) {
        // Prefetch next weight chunk into caches (global_prefetch_b8)
        if (c0 + 32 < Cin)
            __builtin_prefetch(W + ((size_t)mbase) * Cin + c0 + 32, 0, 1);

        for (int tap = 0; tap < KT; ++tap) {
            const int toff = tap - kc;

            // B fragments: X, N = t, K = cin (two b128 loads per lane)
            FragU bf[2];
#pragma unroll
            for (int nf = 0; nf < 2; ++nf) {
                int t = tbase + nf * 16 + ln15 + toff;
                if (t >= 0 && t < T) {
                    const uint4* p =
                        (const uint4*)(Xb + (size_t)t * cinRow + c0 + khalf);
                    bf[nf].q[0] = p[0];   // K = c0+khalf+0..7
                    bf[nf].q[1] = p[2];   // K = c0+khalf+16..23
                } else {
                    bf[nf].q[0] = z4;
                    bf[nf].q[1] = z4;
                }
            }

            const __bf16* Wt = W + ((size_t)tap * Cout) * Cin;
#pragma unroll
            for (int mf = 0; mf < 4; ++mf) {
                int mm = mbase + mf * 16 + ln15;
                const uint4* wp =
                    (const uint4*)(Wt + (size_t)mm * Cin + c0 + khalf);
                FragU af;
                af.q[0] = wp[0];
                af.q[1] = wp[2];
#pragma unroll
                for (int nf = 0; nf < 2; ++nf) {
                    acc[mf][nf] = __builtin_amdgcn_wmma_f32_16x16x32_bf16(
                        false, af.v, false, bf[nf].v,
                        (short)0, acc[mf][nf], false, false);
                }
            }
        }
    }

    // Epilogue: bias + ReLU, pack 8 consecutive cout as one b128 store.
    // C/D layout: elem r, lanes 0-15 -> M=r, N=lane; lanes 16-31 -> M=r+8.
    const int mhi = (lane >> 4) << 3;
#pragma unroll
    for (int nf = 0; nf < 2; ++nf) {
        int t = tbase + nf * 16 + ln15;
        __bf16* yp = Y + ((size_t)b * T + t) * coutRow + coutOff;
#pragma unroll
        for (int mf = 0; mf < 4; ++mf) {
            const int m0 = mbase + mf * 16 + mhi;
            const float4* bp = (const float4*)(bias + m0);
            float4 b0 = bp[0], b1 = bp[1];
            float bv[8] = {b0.x, b0.y, b0.z, b0.w, b1.x, b1.y, b1.z, b1.w};
            OutPack op;
#pragma unroll
            for (int rI = 0; rI < 8; ++rI) {
                float o = acc[mf][nf][rI] + bv[rI];
                if (relu) o = fmaxf(o, 0.f);
                op.b[rI] = (__bf16)o;
            }
            *(uint4*)(yp + m0) = op.q;
        }
    }
}

// ---------------------------------------------------------------------------
// Small head convs (Cout 1 or 2): scalar, f32 weights, bf16 activations.
//   out layout [B][Cout][T] f32 (flat).  mode 1: out = exp(acc * scale[0])
// ---------------------------------------------------------------------------
__global__ void fcos_small_conv(const __bf16* __restrict__ X, int cinRow, int cinOff,
                                const float* __restrict__ W,   // [Cout][Cin][KT]
                                const float* __restrict__ bias,
                                float* __restrict__ out,
                                int Cin, int Cout, int KT, int T, int B,
                                const float* __restrict__ scale, int mode)
{
    int idx = blockIdx.x * blockDim.x + threadIdx.x;
    int total = B * Cout * T;
    if (idx >= total) return;
    int t  = idx % T;
    int co = (idx / T) % Cout;
    int b  = idx / (T * Cout);

    float acc = bias[co];
    int kc = KT >> 1;
    for (int tap = 0; tap < KT; ++tap) {
        int tt = t + tap - kc;
        if (tt < 0 || tt >= T) continue;
        const __bf16* xr = X + ((size_t)b * T + tt) * cinRow + cinOff;
        const float* wr  = W + (size_t)co * Cin * KT + tap;
        float s = 0.f;
        for (int c = 0; c < Cin; ++c)
            s += (float)xr[c] * wr[(size_t)c * KT];
        acc += s;
    }
    out[idx] = (mode == 1) ? expf(acc * scale[0]) : acc;
}

// ---------------------------------------------------------------------------
// Locations: arange(t)*stride + stride/2 per level
// ---------------------------------------------------------------------------
__global__ void fcos_locs_kernel(float* __restrict__ out)
{
    int idx = blockIdx.x * blockDim.x + threadIdx.x;
    if (idx >= 1792) return;
    float v;
    if (idx < 1024)       v = (float)idx * 1.f + 0.5f;
    else if (idx < 1536)  v = (float)(idx - 1024) * 2.f + 1.0f;
    else                  v = (float)(idx - 1536) * 4.f + 2.0f;
    out[idx] = v;
}

// ---------------------------------------------------------------------------
// Host launcher
// ---------------------------------------------------------------------------
static inline void conv_wmma_launch(hipStream_t s,
                                    const __bf16* X, int cinRow, int cinOff,
                                    const __bf16* W, const float* bias,
                                    __bf16* Y, int coutRow, int coutOff,
                                    int Cin, int Cout, int KT, int T, int B, int relu)
{
    dim3 grid((T / 32) / 8, Cout / 64, B);
    fcos_conv_wmma<<<grid, 256, 0, s>>>(X, cinRow, cinOff, W, bias,
                                        Y, coutRow, coutOff,
                                        Cin, Cout, KT, T, relu);
}

extern "C" void kernel_launch(void* const* d_in, const int* in_sizes, int n_in,
                              void* d_out, int out_size, void* d_ws, size_t ws_size,
                              hipStream_t stream)
{
    (void)in_sizes; (void)n_in; (void)out_size; (void)ws_size;
    const int B = 32, C = 512, H = 256;
    const int Ts[3] = {1024, 512, 256};

    const float* x[3] = {(const float*)d_in[0], (const float*)d_in[1], (const float*)d_in[2]};
    const float* cls_w    = (const float*)d_in[3];
    const float* cls_b    = (const float*)d_in[4];
    const float* cls_g    = (const float*)d_in[5];
    const float* cls_beta = (const float*)d_in[6];
    const float* cls_m    = (const float*)d_in[7];
    const float* cls_v    = (const float*)d_in[8];
    const float* box_w    = (const float*)d_in[9];
    const float* box_b    = (const float*)d_in[10];
    const float* box_g    = (const float*)d_in[11];
    const float* box_beta = (const float*)d_in[12];
    const float* box_m    = (const float*)d_in[13];
    const float* box_v    = (const float*)d_in[14];
    const float* logits_w = (const float*)d_in[15];
    const float* logits_b = (const float*)d_in[16];
    const float* pred_w   = (const float*)d_in[17];
    const float* pred_b   = (const float*)d_in[18];
    const float* scales   = (const float*)d_in[19];
    const float* mix_w    = (const float*)d_in[20];
    const float* mix_b    = (const float*)d_in[21];
    const float* mix_g    = (const float*)d_in[22];
    const float* mix_beta = (const float*)d_in[23];
    const float* mix_m    = (const float*)d_in[24];
    const float* mix_v    = (const float*)d_in[25];
    const float* iou_w1   = (const float*)d_in[26];
    const float* iou_b1   = (const float*)d_in[27];
    const float* iou_g    = (const float*)d_in[28];
    const float* iou_beta = (const float*)d_in[29];
    const float* iou_m    = (const float*)d_in[30];
    const float* iou_v    = (const float*)d_in[31];
    const float* iou_w2   = (const float*)d_in[32];
    const float* iou_b2   = (const float*)d_in[33];

    float* dout = (float*)d_out;
    // Output offsets (flat, in return order)
    const int logits_off[3] = {0, 32768, 49152};
    const int bbox_off[3]   = {57344, 122880, 155648};
    const int iou_off[3]    = {172032, 204800, 221184};
    const int locs_off      = 229376;

    // ---- workspace carving ----
    char* ws = (char*)d_ws;
    size_t off = 0;
    auto carve = [&](size_t bytes) -> char* {
        char* p = ws + off;
        off = (off + bytes + 255) & ~(size_t)255;
        return p;
    };
    __bf16* clsWf = (__bf16*)carve((size_t)2 * 3 * C * C * 2);
    float*  clsBf = (float*) carve((size_t)2 * C * 4);
    __bf16* boxWf = (__bf16*)carve((size_t)2 * 3 * C * C * 2);
    float*  boxBf = (float*) carve((size_t)2 * C * 4);
    __bf16* mixWf = (__bf16*)carve((size_t)C * 2 * C * 2);
    float*  mixBf = (float*) carve((size_t)C * 4);
    __bf16* iouWf = (__bf16*)carve((size_t)3 * H * C * 2);
    float*  iouBf = (float*) carve((size_t)H * 4);
    const size_t Tmax = 1024;
    __bf16* xb     = (__bf16*)carve((size_t)B * Tmax * C * 2);
    __bf16* tmp    = (__bf16*)carve((size_t)B * Tmax * C * 2);
    __bf16* mixin  = (__bf16*)carve((size_t)B * Tmax * 2 * C * 2);
    __bf16* mixout = (__bf16*)carve((size_t)B * Tmax * C * 2);
    __bf16* hbuf   = (__bf16*)carve((size_t)B * Tmax * H * 2);

    // ---- fold BN into weights (bf16 repack [tap][cout][cin]) ----
    {
        int n = C * C, blks = (n + 255) / 256;
        for (int l = 0; l < 2; ++l) {
            size_t wo = (size_t)l * C * C * 3;
            fcos_fold_kernel<<<blks, 256, 0, stream>>>(
                cls_w + wo, cls_b + l * C, cls_g + l * C, cls_beta + l * C,
                cls_m + l * C, cls_v + l * C,
                clsWf + (size_t)l * 3 * C * C, clsBf + l * C, C, C, 3);
            fcos_fold_kernel<<<blks, 256, 0, stream>>>(
                box_w + wo, box_b + l * C, box_g + l * C, box_beta + l * C,
                box_m + l * C, box_v + l * C,
                boxWf + (size_t)l * 3 * C * C, boxBf + l * C, C, C, 3);
        }
        int nm = C * 2 * C;
        fcos_fold_kernel<<<(nm + 255) / 256, 256, 0, stream>>>(
            mix_w, mix_b, mix_g, mix_beta, mix_m, mix_v,
            mixWf, mixBf, C, 2 * C, 1);
        int ni = H * C;
        fcos_fold_kernel<<<(ni + 255) / 256, 256, 0, stream>>>(
            iou_w1, iou_b1, iou_g, iou_beta, iou_m, iou_v,
            iouWf, iouBf, H, C, 3);
    }

    // ---- per-level pipeline ----
    for (int l = 0; l < 3; ++l) {
        const int T = Ts[l];

        dim3 tb(32, 8);
        dim3 tg(T / 32, C / 32, B);
        fcos_transpose_cast<<<tg, tb, 0, stream>>>(x[l], xb, C, T);

        // cls tower: xb -> tmp -> mixin[ch 0..511]
        conv_wmma_launch(stream, xb, C, 0, clsWf, clsBf,
                         tmp, C, 0, C, C, 3, T, B, 1);
        conv_wmma_launch(stream, tmp, C, 0, clsWf + (size_t)3 * C * C, clsBf + C,
                         mixin, 2 * C, 0, C, C, 3, T, B, 1);
        // box tower: xb -> tmp -> mixin[ch 512..1023]
        conv_wmma_launch(stream, xb, C, 0, boxWf, boxBf,
                         tmp, C, 0, C, C, 3, T, B, 1);
        conv_wmma_launch(stream, tmp, C, 0, boxWf + (size_t)3 * C * C, boxBf + C,
                         mixin, 2 * C, C, C, C, 3, T, B, 1);
        // mix conv (Cin=1024, K=1) -> mixout
        conv_wmma_launch(stream, mixin, 2 * C, 0, mixWf, mixBf,
                         mixout, C, 0, 2 * C, C, 1, T, B, 1);
        // iou conv1 (512->256, K=3) -> hbuf
        conv_wmma_launch(stream, mixout, C, 0, iouWf, iouBf,
                         hbuf, H, 0, C, H, 3, T, B, 1);

        // logits from ct (mixin ch 0), [B][1][T]
        {
            int n = B * 1 * T;
            fcos_small_conv<<<(n + 255) / 256, 256, 0, stream>>>(
                mixin, 2 * C, 0, logits_w, logits_b, dout + logits_off[l],
                C, 1, 3, T, B, scales, 0);
        }
        // bbox = exp(conv(bt) * scale), [B][2][T]
        {
            int n = B * 2 * T;
            fcos_small_conv<<<(n + 255) / 256, 256, 0, stream>>>(
                mixin, 2 * C, C, pred_w, pred_b, dout + bbox_off[l],
                C, 2, 3, T, B, scales + l, 1);
        }
        // iou head conv2 (256->1, K=1), [B][1][T]
        {
            int n = B * 1 * T;
            fcos_small_conv<<<(n + 255) / 256, 256, 0, stream>>>(
                hbuf, H, 0, iou_w2, iou_b2, dout + iou_off[l],
                H, 1, 1, T, B, scales, 0);
        }
    }

    fcos_locs_kernel<<<(1792 + 255) / 256, 256, 0, stream>>>(dout + locs_off);
}
